// SwinIRBlock_17016660426859
// MI455X (gfx1250) — matrix operand order repaired and checked by reading, not compiled, not run on hardware
//
#include <hip/hip_runtime.h>
#include <hip/hip_bf16.h>

// SwinIR block on gfx1250: bf16 WMMA (v_wmma_f32_16x16x32_bf16) for all GEMMs,
// fused epilogues, windowed layouts, transposed-V for K-contiguous B operands.

typedef __attribute__((ext_vector_type(16))) __bf16 v16bf;
typedef __attribute__((ext_vector_type(8)))  float  v8f;
typedef __attribute__((ext_vector_type(4)))  float  f32x4;
typedef unsigned short u16;

#define DEV static __device__ __forceinline__

DEV u16 f2bf(float f) {
  unsigned u = __float_as_uint(f);
  return (u16)((u + 0x7FFFu + ((u >> 16) & 1u)) >> 16);
}

DEV v8f vzero() {
  v8f z;
#pragma unroll
  for (int i = 0; i < 8; i++) z[i] = 0.f;
  return z;
}

// A fragment (16x32 bf16, MxK): lane holds row M=lane&15; K striping per ISA:
// VGPR j<4: K = 8g+2j,+1 ; j>=4: K = 16+8g+2(j-4),+1  (g = lane>>4)
// p = row base + k0 (element units, 16B aligned)
DEV v16bf ldA(const u16* p, int g) {
  union { f32x4 q[2]; v16bf v; } u;
  u.q[0] = *(const f32x4*)(p + 8 * g);
  u.q[1] = *(const f32x4*)(p + 16 + 8 * g);
  return u.v;
}
// B fragment (32x16 bf16, KxN) from K-contiguous source row (row = output col n):
// same K striping as ldA: elems 0-7 K=8g+s, elems 8-15 K=16+8g+(s-8), column N=lane&15.
DEV v16bf ldB(const u16* p, int g) {
  union { f32x4 q[2]; v16bf v; } u;
  u.q[0] = *(const f32x4*)(p + 8 * g);
  u.q[1] = *(const f32x4*)(p + 16 + 8 * g);
  return u.v;
}

DEV v8f wmma_bf(v16bf a, v16bf b, v8f c) {
  return __builtin_amdgcn_wmma_f32_16x16x32_bf16(false, a, false, b, (short)0, c,
                                                 false, false);
}

DEV float wredsum(float v) {
#pragma unroll
  for (int o = 16; o > 0; o >>= 1) v += __shfl_xor(v, o, 32);
  return v;
}

// ---------------- constants ----------------
#define TOK   262144          // B*H*W
#define SCALEQ 0.17677669529663687f
// workspace offsets (bytes)
#define OFF_WQ    0u
#define OFF_WPROJ 221184u
#define OFF_WFC1  294912u
#define OFF_WFC2  589824u
#define OFF_HB    884736u               // bf16 [TOK][192]    (reused: attnout)
#define OFF_QKV   101548032u            // bf16 [TOK][384]    (reused: hres f32)
#define OFF_VT    302874624u            // bf16 [4096*6][32][64] (reused: mbuf)

// ---------------- kernels ----------------
__global__ void k_cvt(const float* __restrict__ s, u16* __restrict__ d, int n) {
  int i = blockIdx.x * 256 + threadIdx.x;
  if (i < n) d[i] = f2bf(s[i]);
}

// LN1 + cyclic shift + window partition -> hb (windowed row order, bf16)
__global__ void __launch_bounds__(256) k_ln1(const float* __restrict__ x,
                                             const float* __restrict__ gw,
                                             const float* __restrict__ bw,
                                             u16* __restrict__ hb) {
  int lane = threadIdx.x & 31;
  int t = blockIdx.x * 8 + (threadIdx.x >> 5);  // windowed token id
  int win = t >> 6, n = t & 63;
  int b = win >> 8;
  int wy = (win & 255) >> 4, wx = win & 15;
  int sh = (wy * 8 + (n >> 3) + 4) & 127;
  int sw = (wx * 8 + (n & 7) + 4) & 127;
  const float* xr = x + (size_t)(b * 16384 + sh * 128 + sw) * 192;
  float v[6];
  float s = 0.f;
#pragma unroll
  for (int i = 0; i < 6; i++) { v[i] = xr[lane + 32 * i]; s += v[i]; }
  float mu = wredsum(s) * (1.f / 192.f);
  float s2 = 0.f;
#pragma unroll
  for (int i = 0; i < 6; i++) { float d = v[i] - mu; s2 += d * d; }
  float rstd = rsqrtf(wredsum(s2) * (1.f / 192.f) + 1e-5f);
  u16* o = hb + (size_t)t * 192;
#pragma unroll
  for (int i = 0; i < 6; i++) {
    int c = lane + 32 * i;
    o[c] = f2bf((v[i] - mu) * rstd * gw[c] + bw[c]);
  }
}

// QKV GEMM: M=TOK, N=576, K=192. q scaled + row-major, k row-major, v transposed.
__global__ void __launch_bounds__(256) k_qkv(const u16* __restrict__ hb,
                                             const u16* __restrict__ wq,
                                             const float* __restrict__ bias,
                                             u16* __restrict__ qk,
                                             u16* __restrict__ vT) {
  int lane = threadIdx.x & 31, w = threadIdx.x >> 5;
  int r = lane & 15, g = lane >> 4;
  int mbase = blockIdx.x * 128 + (w >> 1) * 32;
  int nbase = blockIdx.y * 64 + (w & 1) * 32;
  v8f acc[2][2];
  acc[0][0] = vzero(); acc[0][1] = vzero(); acc[1][0] = vzero(); acc[1][1] = vzero();
  int row0 = mbase + r, row1 = mbase + 16 + r;
  int nr0 = nbase + r, nr1 = nbase + 16 + r;
#pragma unroll
  for (int ks = 0; ks < 6; ks++) {
    int k0 = ks * 32;
    v16bf a0 = ldA(hb + (size_t)row0 * 192 + k0, g);
    v16bf a1 = ldA(hb + (size_t)row1 * 192 + k0, g);
    v16bf b0 = ldB(wq + (size_t)nr0 * 192 + k0, g);
    v16bf b1 = ldB(wq + (size_t)nr1 * 192 + k0, g);
    acc[0][0] = wmma_bf(a0, b0, acc[0][0]);
    acc[0][1] = wmma_bf(a0, b1, acc[0][1]);
    acc[1][0] = wmma_bf(a1, b0, acc[1][0]);
    acc[1][1] = wmma_bf(a1, b1, acc[1][1]);
  }
#pragma unroll
  for (int mt = 0; mt < 2; mt++)
#pragma unroll
    for (int nt = 0; nt < 2; nt++) {
      int c0 = nbase + nt * 16;
      int col = c0 + r;
      float bval = bias[col];
#pragma unroll
      for (int vi = 0; vi < 8; vi++) {
        int row = mbase + mt * 16 + vi + 8 * g;
        float val = acc[mt][nt][vi] + bval;
        if (c0 < 192) {
          qk[(size_t)row * 384 + col] = f2bf(val * SCALEQ);
        } else if (c0 < 384) {
          qk[(size_t)row * 384 + col] = f2bf(val);
        } else {
          int f = col - 384;
          int hh = f >> 5, dd = f & 31;
          int win = row >> 6, key = row & 63;
          vT[(size_t)((win * 6 + hh) * 32 + dd) * 64 + key] = f2bf(val);
        }
      }
    }
}

// Attention: 1 block/window, wave==head. S=q*k^T (+relpos bias +shift mask),
// LDS softmax (in-place bf16 P), then P*V from transposed V.
__global__ void __launch_bounds__(192) k_attn(const u16* __restrict__ qk,
                                              const u16* __restrict__ vT,
                                              const float* __restrict__ rpb,
                                              u16* __restrict__ ao) {
  extern __shared__ char smem[];
  float* S = (float*)smem;  // [6][64][68] (68 pad: bank-safe + 16B rows)
  int lane = threadIdx.x & 31, h = threadIdx.x >> 5;
  int r = lane & 15, g = lane >> 4;
  int win = blockIdx.x;
  int wy = (win & 255) >> 4, wx = win & 15;
  float* Sh = S + h * 64 * 68;
  const u16* qwin = qk + (size_t)win * 64 * 384;

  v16bf bf[4];
#pragma unroll
  for (int nt = 0; nt < 4; nt++)
    bf[nt] = ldB(qwin + (size_t)(nt * 16 + r) * 384 + 192 + h * 32, g);

#pragma unroll
  for (int mt = 0; mt < 4; mt++) {
    v16bf a = ldA(qwin + (size_t)(mt * 16 + r) * 384 + h * 32, g);
#pragma unroll
    for (int nt = 0; nt < 4; nt++) {
      v8f acc = vzero();
      acc = wmma_bf(a, bf[nt], acc);
#pragma unroll
      for (int vi = 0; vi < 8; vi++) {
        int nq = mt * 16 + vi + 8 * g;
        int nk = nt * 16 + r;
        int iy = nq >> 3, ix = nq & 7, jy = nk >> 3, jx = nk & 7;
        float bias = rpb[((iy - jy + 7) * 15 + (ix - jx + 7)) * 6 + h];
        int ah = wy * 8 + iy, aw = wx * 8 + ix;
        int bh = wy * 8 + jy, bw2 = wx * 8 + jx;
        int la = (ah < 120 ? 0 : (ah < 124 ? 1 : 2)) * 3 + (aw < 120 ? 0 : (aw < 124 ? 1 : 2));
        int lb = (bh < 120 ? 0 : (bh < 124 ? 1 : 2)) * 3 + (bw2 < 120 ? 0 : (bw2 < 124 ? 1 : 2));
        float mval = (la != lb) ? -100.f : 0.f;
        Sh[nq * 68 + nk] = acc[vi] + bias + mval;
      }
    }
  }
  __syncthreads();
#pragma unroll
  for (int rr = 0; rr < 2; rr++) {
    float* row = Sh + (lane + 32 * rr) * 68;
    u16* prow = (u16*)row;
    float mx = -1e30f;
    for (int j = 0; j < 64; j++) mx = fmaxf(mx, row[j]);
    float sum = 0.f;
    for (int j = 0; j < 64; j++) { float e = __expf(row[j] - mx); row[j] = e; sum += e; }
    float inv = 1.f / sum;
    for (int j = 0; j < 64; j++) prow[j] = f2bf(row[j] * inv);  // in-place (safe: write slot j/2 <= read slot j)
  }
  __syncthreads();
  const u16* vbase = vT + (size_t)(win * 6 + h) * 32 * 64;
#pragma unroll
  for (int mt = 0; mt < 4; mt++) {
    const u16* prow = (const u16*)(Sh + (mt * 16 + r) * 68);
    v16bf a0 = ldA(prow, g);
    v16bf a1 = ldA(prow + 32, g);
#pragma unroll
    for (int nt = 0; nt < 2; nt++) {
      v16bf b0 = ldB(vbase + (size_t)(nt * 16 + r) * 64, g);
      v16bf b1 = ldB(vbase + (size_t)(nt * 16 + r) * 64 + 32, g);
      v8f acc = vzero();
      acc = wmma_bf(a0, b0, acc);
      acc = wmma_bf(a1, b1, acc);
#pragma unroll
      for (int vi = 0; vi < 8; vi++) {
        int rowg = win * 64 + mt * 16 + vi + 8 * g;
        int col = h * 32 + nt * 16 + r;
        ao[(size_t)rowg * 192 + col] = f2bf(acc[vi]);
      }
    }
  }
}

// proj GEMM (window-reverse gather on A rows) + residual + LN2 -> hres(f32), m(bf16)
__global__ void __launch_bounds__(256) k_proj(const u16* __restrict__ ao,
                                              const u16* __restrict__ wp,
                                              const float* __restrict__ pb,
                                              const float* __restrict__ x,
                                              const float* __restrict__ g2,
                                              const float* __restrict__ b2,
                                              float* __restrict__ hres,
                                              u16* __restrict__ mbuf) {
  extern __shared__ char smem[];
  float* Hs = (float*)smem;       // [64][196] padded
  float* mus = Hs + 64 * 196;     // [64]
  float* rss = mus + 64;          // [64]
  int lane = threadIdx.x & 31, w = threadIdx.x >> 5;
  int r = lane & 15, g = lane >> 4;
  int mbase = blockIdx.x * 64;
  int mt = w & 3, ng = w >> 2;
  // window-reverse + un-shift permuted source row for this lane's A rows
  int t = mbase + mt * 16 + r;
  int b = t >> 14, pix = t & 16383;
  int hh = pix >> 7, ww = pix & 127;
  int hs = (hh + 124) & 127, ws2 = (ww + 124) & 127;
  int srow = b * 16384 + ((hs >> 3) * 16 + (ws2 >> 3)) * 64 + (hs & 7) * 8 + (ws2 & 7);
  const u16* abase = ao + (size_t)srow * 192;
  v8f acc[6];
#pragma unroll
  for (int j = 0; j < 6; j++) acc[j] = vzero();
#pragma unroll
  for (int ks = 0; ks < 6; ks++) {
    v16bf a = ldA(abase + ks * 32, g);
#pragma unroll
    for (int j = 0; j < 6; j++) {
      int nrow = (ng * 6 + j) * 16 + r;
      v16bf bb = ldB(wp + (size_t)nrow * 192 + ks * 32, g);
      acc[j] = wmma_bf(a, bb, acc[j]);
    }
  }
#pragma unroll
  for (int j = 0; j < 6; j++) {
    int c = (ng * 6 + j) * 16 + r;
    float bv = pb[c];
#pragma unroll
    for (int vi = 0; vi < 8; vi++) {
      int rowl = mt * 16 + vi + 8 * g;
      int tt = mbase + rowl;
      float val = acc[j][vi] + bv + x[(size_t)tt * 192 + c];
      Hs[rowl * 196 + c] = val;
      hres[(size_t)tt * 192 + c] = val;
    }
  }
  __syncthreads();
  if (threadIdx.x < 64) {
    const float* rrow = Hs + threadIdx.x * 196;
    float s = 0.f;
    for (int c = 0; c < 192; c++) s += rrow[c];
    float mu = s * (1.f / 192.f);
    float s2 = 0.f;
    for (int c = 0; c < 192; c++) { float d = rrow[c] - mu; s2 += d * d; }
    mus[threadIdx.x] = mu;
    rss[threadIdx.x] = rsqrtf(s2 * (1.f / 192.f) + 1e-5f);
  }
  __syncthreads();
  for (int i = threadIdx.x; i < 64 * 192; i += 256) {
    int rowl = i / 192, c = i - rowl * 192;
    float val = (Hs[rowl * 196 + c] - mus[rowl]) * rss[rowl] * g2[c] + b2[c];
    mbuf[(size_t)(mbase + rowl) * 192 + c] = f2bf(val);
  }
}

// fused MLP: fc1+GELU -> LDS (64x768 bf16), fc2 + bias + residual -> out (f32)
__global__ void __launch_bounds__(256) k_mlp(const u16* __restrict__ mbuf,
                                             const u16* __restrict__ w1,
                                             const float* __restrict__ b1v,
                                             const u16* __restrict__ w2,
                                             const float* __restrict__ b2v,
                                             const float* __restrict__ hres,
                                             float* __restrict__ out) {
  extern __shared__ char smem[];
  u16* G1 = (u16*)smem;  // [64][776] padded (16B-aligned rows, bank-skewed)
  int lane = threadIdx.x & 31, w = threadIdx.x >> 5;
  int r = lane & 15, g = lane >> 4;
  int mbase = blockIdx.x * 64;
  int mt = w & 3, half = w >> 2;
  const u16* arow = mbuf + (size_t)(mbase + mt * 16 + r) * 192;
  v16bf aF[6];
#pragma unroll
  for (int ks = 0; ks < 6; ks++) aF[ks] = ldA(arow + ks * 32, g);
  for (int ch = 0; ch < 4; ch++) {
    int ntb = half * 24 + ch * 6;
    v8f acc[6];
#pragma unroll
    for (int j = 0; j < 6; j++) acc[j] = vzero();
#pragma unroll
    for (int ks = 0; ks < 6; ks++) {
#pragma unroll
      for (int j = 0; j < 6; j++) {
        int nrow = (ntb + j) * 16 + r;
        v16bf bb = ldB(w1 + (size_t)nrow * 192 + ks * 32, g);
        acc[j] = wmma_bf(aF[ks], bb, acc[j]);
      }
    }
#pragma unroll
    for (int j = 0; j < 6; j++) {
      int c = (ntb + j) * 16 + r;
      float bv = b1v[c];
#pragma unroll
      for (int vi = 0; vi < 8; vi++) {
        int rowl = mt * 16 + vi + 8 * g;
        float xv = acc[j][vi] + bv;
        float ge = 0.5f * xv * (1.f + erff(xv * 0.7071067811865475f));
        G1[rowl * 776 + c] = f2bf(ge);
      }
    }
  }
  __syncthreads();
  v8f accB[6];
#pragma unroll
  for (int j = 0; j < 6; j++) accB[j] = vzero();
  const u16* g1row = G1 + (mt * 16 + r) * 776;
  for (int ks = 0; ks < 24; ks++) {
    v16bf a = ldA(g1row + ks * 32, g);
#pragma unroll
    for (int j = 0; j < 6; j++) {
      int nrow = (half * 6 + j) * 16 + r;
      v16bf bb = ldB(w2 + (size_t)nrow * 768 + ks * 32, g);
      accB[j] = wmma_bf(a, bb, accB[j]);
    }
  }
#pragma unroll
  for (int j = 0; j < 6; j++) {
    int c = (half * 6 + j) * 16 + r;
    float bv = b2v[c];
#pragma unroll
    for (int vi = 0; vi < 8; vi++) {
      int tt = mbase + mt * 16 + vi + 8 * g;
      out[(size_t)tt * 192 + c] = accB[j][vi] + bv + hres[(size_t)tt * 192 + c];
    }
  }
}

extern "C" void kernel_launch(void* const* d_in, const int* in_sizes, int n_in,
                              void* d_out, int out_size, void* d_ws, size_t ws_size,
                              hipStream_t stream) {
  const float* x     = (const float*)d_in[0];
  const float* n1g   = (const float*)d_in[1];
  const float* n1b   = (const float*)d_in[2];
  const float* qkvw  = (const float*)d_in[3];
  const float* qkvb  = (const float*)d_in[4];
  const float* projw = (const float*)d_in[5];
  const float* projb = (const float*)d_in[6];
  const float* rpb   = (const float*)d_in[7];
  const float* n2g   = (const float*)d_in[8];
  const float* n2b   = (const float*)d_in[9];
  const float* fc1w  = (const float*)d_in[10];
  const float* fc1b  = (const float*)d_in[11];
  const float* fc2w  = (const float*)d_in[12];
  const float* fc2b  = (const float*)d_in[13];
  float* out = (float*)d_out;
  char* ws = (char*)d_ws;

  u16* wq    = (u16*)(ws + OFF_WQ);
  u16* wp    = (u16*)(ws + OFF_WPROJ);
  u16* w1    = (u16*)(ws + OFF_WFC1);
  u16* w2    = (u16*)(ws + OFF_WFC2);
  u16* hb    = (u16*)(ws + OFF_HB);     // later reused as attnout
  u16* qk    = (u16*)(ws + OFF_QKV);    // later reused as hres (f32)
  u16* vT    = (u16*)(ws + OFF_VT);     // later reused as mbuf
  u16* ao    = hb;
  float* hres = (float*)(ws + OFF_QKV);
  u16* mbuf  = vT;

  k_cvt<<<(110592 + 255) / 256, 256, 0, stream>>>(qkvw, wq, 110592);
  k_cvt<<<(36864 + 255) / 256, 256, 0, stream>>>(projw, wp, 36864);
  k_cvt<<<(147456 + 255) / 256, 256, 0, stream>>>(fc1w, w1, 147456);
  k_cvt<<<(147456 + 255) / 256, 256, 0, stream>>>(fc2w, w2, 147456);

  k_ln1<<<TOK / 8, 256, 0, stream>>>(x, n1g, n1b, hb);
  k_qkv<<<dim3(TOK / 128, 9), 256, 0, stream>>>(hb, wq, qkvb, qk, vT);
  k_attn<<<4096, 192, 6 * 64 * 68 * sizeof(float), stream>>>(qk, vT, rpb, ao);
  k_proj<<<TOK / 64, 256, (64 * 196 + 128) * sizeof(float), stream>>>(
      ao, wp, projb, x, n2g, n2b, hres, mbuf);
  k_mlp<<<TOK / 64, 256, 64 * 776 * sizeof(u16), stream>>>(
      mbuf, w1, fc1b, w2, fc2b, hres, out);
}
